// RNN_3848290697662
// MI455X (gfx1250) — matrix operand order repaired and checked by reading, not compiled, not run on hardware
//
#include <hip/hip_runtime.h>

#define SEQ   512
#define BATCH 128
#define EMB   256
#define HID   1024

#define SCAN_BLOCKS 64          // 64 blocks x 8 waves = 512 waves = 512 16x16 tiles
#define USE_ASYNC_LDS 1

typedef __attribute__((ext_vector_type(16))) __bf16        bf16x16;
typedef __attribute__((ext_vector_type(8)))  float         f32x8;
typedef __attribute__((ext_vector_type(4)))  unsigned int  u32x4;

union FragB16 {
    bf16x16        v;
    u32x4          q[2];
    unsigned short us[16];
};

__device__ __forceinline__ unsigned short f32_to_bf16(float f) {
    union { float f; unsigned int u; } c;
    c.f = f;
    unsigned int u = c.u;
    u += 0x7FFFu + ((u >> 16) & 1u);   // round-to-nearest-even
    return (unsigned short)(u >> 16);
}

// A fragment (16x32 bf16) from row-major f32 src with on-the-fly conversion.
// ISA 7.12.2: lane<16 holds K {0..7,16..23}; lane>=16 holds K {8..15,24..31}.
__device__ __forceinline__ void load_a_f32(FragB16& f, const float* __restrict__ src,
                                           int ld, int row0, int k0, int lane) {
    const int m    = row0 + (lane & 15);
    const int half = lane >> 4;
    const float* p = src + (size_t)m * ld + k0 + half * 8;
#pragma unroll
    for (int j = 0; j < 8; ++j) {
        f.us[j]     = f32_to_bf16(p[j]);
        f.us[8 + j] = f32_to_bf16(p[16 + j]);
    }
}

// A fragment (16x32 bf16) from an LDS-resident 16xHID strip (row-major).
__device__ __forceinline__ void load_a_lds(FragB16& f, const unsigned short* As,
                                           int k0, int lane) {
    const int m    = lane & 15;
    const int half = lane >> 4;
    const unsigned short* p = As + m * HID + k0 + half * 8;
    f.q[0] = *reinterpret_cast<const u32x4*>(p);        // K = half*8 + 0..7
    f.q[1] = *reinterpret_cast<const u32x4*>(p + 16);   // K = 16 + half*8 + 0..7
}

// B fragment (32x16 bf16) from N-major (transposed) weights Wt[n][k].
// ISA 7.12.2: lane<16 holds K {0..7,16..23}; lane>=16 holds K {8..15,24..31}; lane%16 = n.
__device__ __forceinline__ void load_b_bf16(FragB16& f, const unsigned short* __restrict__ Wt,
                                            int ld, int n0, int k0, int lane) {
    const int n    = n0 + (lane & 15);
    const int half = lane >> 4;
    const unsigned short* p = Wt + (size_t)n * ld + k0 + half * 8;
    f.q[0] = *reinterpret_cast<const u32x4*>(p);       // K = half*8 + 0..7
    f.q[1] = *reinterpret_cast<const u32x4*>(p + 16);  // K = 16 + half*8 + 0..7
}

// Device-wide barrier for the persistent scan kernel (all SCAN_BLOCKS resident).
__device__ __forceinline__ void grid_sync(int* count, int* gen) {
    __syncthreads();
    __threadfence();
    if (threadIdx.x == 0) {
        const int g = __hip_atomic_load(gen, __ATOMIC_RELAXED, __HIP_MEMORY_SCOPE_AGENT);
        if (__hip_atomic_fetch_add(count, 1, __ATOMIC_ACQ_REL, __HIP_MEMORY_SCOPE_AGENT)
                == SCAN_BLOCKS - 1) {
            __hip_atomic_store(count, 0, __ATOMIC_RELAXED, __HIP_MEMORY_SCOPE_AGENT);
            __hip_atomic_fetch_add(gen, 1, __ATOMIC_RELEASE, __HIP_MEMORY_SCOPE_AGENT);
        } else {
            while (__hip_atomic_load(gen, __ATOMIC_ACQUIRE, __HIP_MEMORY_SCOPE_AGENT) == g) {
                __builtin_amdgcn_s_sleep(1);
            }
        }
    }
    __syncthreads();
}

// One-time prep: W_h -> bf16 transposed, W_in -> bf16 transposed, h0 -> bf16,
// and reset the grid-barrier state (determinism under graph replay).
__global__ void k_prep(const float* __restrict__ Wh, const float* __restrict__ Win,
                       const float* __restrict__ h0f,
                       unsigned short* __restrict__ Wh_t, unsigned short* __restrict__ Wi_t,
                       unsigned short* __restrict__ h0, int* __restrict__ bar) {
    const int idx = blockIdx.x * blockDim.x + threadIdx.x;
    if (idx < HID * HID) {
        const int k = idx / HID, n = idx % HID;
        Wh_t[(size_t)n * HID + k] = f32_to_bf16(Wh[idx]);
    }
    if (idx < EMB * HID) {
        const int e = idx / HID, n = idx % HID;
        Wi_t[(size_t)n * EMB + e] = f32_to_bf16(Win[idx]);
    }
    if (idx < BATCH * HID) {
        h0[idx] = f32_to_bf16(h0f[idx]);
    }
    if (idx < 2) {
        bar[idx] = 0;
    }
}

// Phase 1: xp = x @ W_in + b_in + b_h, written straight into d_out [S*B, HID].
__global__ void k_inproj(const float* __restrict__ x, const unsigned short* __restrict__ Wi_t,
                         const float* __restrict__ b_in, const float* __restrict__ b_h,
                         float* __restrict__ out) {
    const int lane   = threadIdx.x & 31;
    const int wave   = threadIdx.x >> 5;
    const int waveId = blockIdx.x * 8 + wave;
    const int nTiles = HID / 16;                 // 64
    const int m0     = (waveId / nTiles) * 16;   // row tile in S*B
    const int n0     = (waveId % nTiles) * 16;   // col tile in HID

    f32x8  acc = {};
    FragB16 a, b;
#pragma unroll
    for (int k0 = 0; k0 < EMB; k0 += 32) {
        load_a_f32(a, x, EMB, m0, k0, lane);
        load_b_bf16(b, Wi_t, EMB, n0, k0, lane);
        acc = __builtin_amdgcn_wmma_f32_16x16x32_bf16(false, a.v, false, b.v,
                                                      (short)0, acc, false, false);
    }

    const int   n    = n0 + (lane & 15);
    const int   half = lane >> 4;
    const float bias = b_in[n] + b_h[n];
#pragma unroll
    for (int r = 0; r < 8; ++r) {
        const int row = m0 + half * 8 + r;
        out[(size_t)row * HID + n] = acc[r] + bias;
    }
}

// Phase 2: persistent scan kernel. One launch runs all 512 timesteps:
//   h = tanh(xp + h_prev @ W_h)
// Per step: stage this block's 16xHID strip of h_prev into LDS (async B128 copies,
// shared by the 8 waves), 32x WMMA over K=1024 with B-fragments from the
// L2-resident transposed W_h, tanh epilogue, then device-wide barrier.
__global__ void k_scan(const unsigned short* __restrict__ Wh_t,
                       float* __restrict__ out,
                       unsigned short* __restrict__ hbuf0,
                       unsigned short* __restrict__ hbuf1,
                       int* __restrict__ bar) {
    __shared__ unsigned short As[16 * HID];      // 32 KB A strip (batch rows m0..m0+15)

    const int lane   = threadIdx.x & 31;
    const int wave   = threadIdx.x >> 5;
    const int waveId = blockIdx.x * 8 + wave;
    const int nTiles = HID / 16;                 // 64
    const int m0     = (waveId / nTiles) * 16;   // batch tile (blocks 0..7 share m0=0, etc.)
    const int n0     = (waveId % nTiles) * 16;   // hidden tile

    const int n    = n0 + (lane & 15);
    const int half = lane >> 4;
    const unsigned short* wrow = Wh_t + (size_t)n * HID;

    for (int s = 0; s < SEQ; ++s) {
        const unsigned short* __restrict__ h_in  = (s & 1) ? hbuf1 : hbuf0;
        unsigned short*       __restrict__ h_out = (s & 1) ? hbuf0 : hbuf1;
        float* __restrict__ out_s = out + (size_t)s * BATCH * HID;

        // ---- Stage A strip h_in[m0 .. m0+15][0 .. HID) into LDS -------------
        {
            const unsigned short* src = h_in + (size_t)m0 * HID;
            // 32 KB = 2048 x 16B chunks; 256 threads x 8 chunks.
#pragma unroll
            for (int c = 0; c < 8; ++c) {
                const int i = (threadIdx.x + c * 256) * 8;   // ushort index, 16B aligned
#if USE_ASYNC_LDS
                unsigned ldsAddr = (unsigned)(uintptr_t)(&As[i]);
                asm volatile("global_load_async_to_lds_b128 %0, %1, off"
                             :: "v"(ldsAddr), "v"(src + i) : "memory");
#else
                *reinterpret_cast<u32x4*>(&As[i]) =
                    *reinterpret_cast<const u32x4*>(src + i);
#endif
            }
#if USE_ASYNC_LDS
            asm volatile("s_wait_asynccnt 0" ::: "memory");
#endif
            __syncthreads();
        }

        // ---- 16x16 tile GEMM: K = 1024 in chunks of 32 ----------------------
        f32x8  acc = {};
        FragB16 a, b;
#pragma unroll 4
        for (int k0 = 0; k0 < HID; k0 += 32) {
            __builtin_prefetch(wrow + k0 + 64, 0, 0);   // global_prefetch_b8: next W_h chunk
            load_a_lds(a, As, k0, lane);                // ds_load_b128 from staged strip
            load_b_bf16(b, Wh_t, HID, n0, k0, lane);
            acc = __builtin_amdgcn_wmma_f32_16x16x32_bf16(false, a.v, false, b.v,
                                                          (short)0, acc, false, false);
        }

        // ---- Epilogue: h = tanh(xp + acc); fp32 to out, bf16 to next state --
#pragma unroll
        for (int r = 0; r < 8; ++r) {
            const int    bi  = m0 + half * 8 + r;
            const size_t idx = (size_t)bi * HID + n;
            const float  v   = tanhf(out_s[idx] + acc[r]);
            out_s[idx] = v;                 // final output h_seq[s]
            h_out[idx] = f32_to_bf16(v);    // recurrent state for step s+1
        }

        // LDS strip is re-written next iteration; barrier covers both LDS reuse
        // and global h_out visibility across blocks.
        grid_sync(bar, bar + 1);
    }
}

extern "C" void kernel_launch(void* const* d_in, const int* in_sizes, int n_in,
                              void* d_out, int out_size, void* d_ws, size_t ws_size,
                              hipStream_t stream) {
    (void)in_sizes; (void)n_in; (void)out_size; (void)ws_size;
    const float* x    = (const float*)d_in[0];
    const float* h0   = (const float*)d_in[1];
    const float* W_in = (const float*)d_in[2];
    const float* b_in = (const float*)d_in[3];
    const float* W_h  = (const float*)d_in[4];
    const float* b_h  = (const float*)d_in[5];
    float* out = (float*)d_out;

    // Workspace layout (bf16 as ushort + barrier ints): ~3 MB total.
    unsigned short* Wh_t  = (unsigned short*)d_ws;            // HID*HID
    unsigned short* Wi_t  = Wh_t  + (size_t)HID * HID;        // HID*EMB
    unsigned short* hbuf0 = Wi_t  + (size_t)HID * EMB;        // BATCH*HID
    unsigned short* hbuf1 = hbuf0 + (size_t)BATCH * HID;      // BATCH*HID
    int*            bar   = (int*)(hbuf1 + (size_t)BATCH * HID);

    // 1) Convert weights (transposed for contiguous B-fragment loads),
    //    initial h, and reset the grid barrier.
    k_prep<<<(HID * HID + 255) / 256, 256, 0, stream>>>(W_h, W_in, h0, Wh_t, Wi_t, hbuf0, bar);

    // 2) Input projection GEMM into d_out: 262144 16x16 tiles, 8 waves/block.
    k_inproj<<<(SEQ * BATCH / 16) * (HID / 16) / 8, 256, 0, stream>>>(x, Wi_t, b_in, b_h, out);

    // 3) Entire 512-step recurrence in ONE persistent kernel (64 resident blocks,
    //    device-wide barrier between steps) -- removes 512 launch overheads.
    k_scan<<<SCAN_BLOCKS, 256, 0, stream>>>(Wh_t, out, hbuf0, hbuf1, bar);
}
